// DrBCNet_21500606284502
// MI455X (gfx1250) — matrix operand order repaired and checked by reading, not compiled, not run on hardware
//
#include <hip/hip_runtime.h>

#define N_NODES 30000
#define N_EDGES 600000
#define H 128
#define TH3 384
#define NLAYERS 5
#define EPSN 1e-12f

typedef unsigned short u16;
typedef unsigned int u32;
typedef __attribute__((ext_vector_type(16))) __bf16 v16bf;
typedef __attribute__((ext_vector_type(8)))  float  v8f;

union ABu { uint4 q[2]; v16bf v; };

__device__ __forceinline__ u16 f2bf(float f) {
    u32 u = __float_as_uint(f);
    u32 r = u + 0x7FFFu + ((u >> 16) & 1u);   // round-to-nearest-even
    return (u16)(r >> 16);
}

__device__ __forceinline__ v8f wmma_bf16(v16bf a, v16bf b, v8f c) {
    return __builtin_amdgcn_wmma_f32_16x16x32_bf16(false, a, false, b, (short)0, c, false, false);
}

__device__ __forceinline__ float sigm(float x) { return 1.0f / (1.0f + expf(-x)); }

// ---------------- utility kernels ----------------
__global__ __launch_bounds__(256) void zero_f32(float* __restrict__ p, int n) {
    int i = blockIdx.x * 256 + threadIdx.x;
    if (i < n) p[i] = 0.0f;
}

__global__ __launch_bounds__(256) void cvt_f2bf(const float* __restrict__ s, u16* __restrict__ d, int n) {
    int i = blockIdx.x * 256 + threadIdx.x;
    if (i < n) d[i] = f2bf(s[i]);
}

// ---------------- encoder: h = l2norm(relu(x @ W1^T + b1)) ----------------
__global__ __launch_bounds__(128) void encoder_kernel(
    const float* __restrict__ x, const float* __restrict__ W1, const float* __restrict__ b1,
    float* __restrict__ h, u16* __restrict__ h_bf,
    float* __restrict__ hmax, u16* __restrict__ hmax_bf)
{
    int node = blockIdx.x;
    int t = threadIdx.x;
    float x0 = x[node * 3 + 0], x1 = x[node * 3 + 1], x2 = x[node * 3 + 2];
    float v = fmaf(W1[t * 3 + 0], x0, fmaf(W1[t * 3 + 1], x1, fmaf(W1[t * 3 + 2], x2, b1[t])));
    v = fmaxf(v, 0.0f);
    __shared__ float ss;
    if (t == 0) ss = 0.0f;
    __syncthreads();
    atomicAdd(&ss, v * v);
    __syncthreads();
    float val = v / (sqrtf(ss) + EPSN);
    int o = node * H + t;
    h[o] = val;          h_bf[o] = f2bf(val);
    hmax[o] = val;       hmax_bf[o] = f2bf(val);
}

// ---------------- edge scatter: agg[dst] += h[src] ----------------
__global__ __launch_bounds__(256) void scatter_kernel(
    const float* __restrict__ h, const int* __restrict__ esrc, const int* __restrict__ edst,
    float* __restrict__ agg)
{
    int wave = threadIdx.x >> 5;
    int lane = threadIdx.x & 31;
    int e = blockIdx.x * 8 + wave;
    int s = esrc[e], d = edst[e];
    int k = lane * 4;
    const float4 hv = *(const float4*)(h + s * H + k);
    float* ap = agg + d * H + k;
    __hip_atomic_fetch_add(ap + 0, hv.x, __ATOMIC_RELAXED, __HIP_MEMORY_SCOPE_AGENT);
    __hip_atomic_fetch_add(ap + 1, hv.y, __ATOMIC_RELAXED, __HIP_MEMORY_SCOPE_AGENT);
    __hip_atomic_fetch_add(ap + 2, hv.z, __ATOMIC_RELAXED, __HIP_MEMORY_SCOPE_AGENT);
    __hip_atomic_fetch_add(ap + 3, hv.w, __ATOMIC_RELAXED, __HIP_MEMORY_SCOPE_AGENT);
}

// ---------------- fused GRU layer: WMMA GEMMs + gates + l2norm + hmax ----------------
// 1 block = 16-node tile, 256 threads = 8 waves.
// 48 jobs: 24 column tiles x {gi (A=agg_bf, W=Wih), gh (A=h_bf, W=Whh)}; 6 jobs per wave.
__global__ __launch_bounds__(256) void layer_kernel(
    const u16* __restrict__ agg_bf, u16* __restrict__ h_bf,
    const u16* __restrict__ Wih_bf, const u16* __restrict__ Whh_bf,
    const float* __restrict__ b_ih, const float* __restrict__ b_hh,
    float* __restrict__ h, float* __restrict__ hmax, u16* __restrict__ hmax_bf)
{
    __shared__ float gi_s[16 * TH3];
    __shared__ float gh_s[16 * TH3];
    __shared__ float norm_s[16];

    int tid  = threadIdx.x;
    int lane = tid & 31;
    int wave = tid >> 5;
    int row  = lane & 15;      // A-row within tile == D-column within col-tile
    int half = lane >> 4;
    int tile = blockIdx.x;
    int nodeRow = tile * 16 + row;

    const u16* ArowGi = agg_bf + nodeRow * H;
    const u16* ArowGh = h_bf   + nodeRow * H;

    #pragma unroll
    for (int jj = 0; jj < 6; ++jj) {
        int job   = wave * 6 + jj;
        int which = job & 1;          // 0: gi, 1: gh
        int ct    = job >> 1;         // column tile 0..23
        int jBase = ct * 16;
        const u16* Arow = which ? ArowGh : ArowGi;
        const u16* Brow = (which ? Whh_bf : Wih_bf) + (jBase + row) * H;

        v8f c = {0.f, 0.f, 0.f, 0.f, 0.f, 0.f, 0.f, 0.f};
        #pragma unroll
        for (int kk = 0; kk < 4; ++kk) {
            int ks = kk * 32;
            ABu a, b;
            // A 16x32 bf16: lane half0 -> K {ks..ks+7, ks+16..ks+23}; half1 -> +8
            a.q[0] = *(const uint4*)(Arow + ks + half * 8);
            a.q[1] = *(const uint4*)(Arow + ks + 16 + half * 8);
            // B 32x16 bf16: lane half0 -> K {ks..ks+7, ks+16..ks+23}; half1 -> +8
            b.q[0] = *(const uint4*)(Brow + ks + half * 8);
            b.q[1] = *(const uint4*)(Brow + ks + 16 + half * 8);
            c = wmma_bf16(a.v, b.v, c);
        }
        float* outS = which ? gh_s : gi_s;
        int colD = jBase + row;
        float bv = which ? b_hh[colD] : b_ih[colD];
        #pragma unroll
        for (int v = 0; v < 8; ++v) {
            int m = v + 8 * half;
            outS[m * TH3 + colD] = c[v] + bv;
        }
    }
    __syncthreads();
    if (tid < 16) norm_s[tid] = 0.0f;
    __syncthreads();

    // GRU gates; stash h_new into gi_s[m*TH3 + k] (i_r slot already consumed per-thread)
    #pragma unroll
    for (int idx = tid; idx < 16 * H; idx += 256) {
        int m = idx >> 7;
        int k = idx & (H - 1);
        int node = tile * 16 + m;
        int g = m * TH3;
        float i_r = gi_s[g + k], i_z = gi_s[g + H + k], i_n = gi_s[g + 2 * H + k];
        float h_r = gh_s[g + k], h_z = gh_s[g + H + k], h_n = gh_s[g + 2 * H + k];
        float r = sigm(i_r + h_r);
        float z = sigm(i_z + h_z);
        float nn = tanhf(i_n + r * h_n);
        float hold = h[node * H + k];
        float hv = (1.0f - z) * nn + z * hold;
        gi_s[g + k] = hv;
        atomicAdd(&norm_s[m], hv * hv);
    }
    __syncthreads();

    // normalize, write h / h_bf / hmax / hmax_bf
    #pragma unroll
    for (int idx = tid; idx < 16 * H; idx += 256) {
        int m = idx >> 7;
        int k = idx & (H - 1);
        int node = tile * 16 + m;
        float inv = 1.0f / (sqrtf(norm_s[m]) + EPSN);
        float val = gi_s[m * TH3 + k] * inv;
        int o = node * H + k;
        h[o] = val;
        h_bf[o] = f2bf(val);
        float hm = fmaxf(hmax[o], val);
        hmax[o] = hm;
        hmax_bf[o] = f2bf(hm);
    }
}

// ---------------- decoder: out = hmax @ W2^T + b2 ----------------
// 1 block = 16-node tile, 8 waves = 8 column tiles of 16.
__global__ __launch_bounds__(256) void decoder_kernel(
    const u16* __restrict__ hmax_bf, const u16* __restrict__ W2_bf,
    const float* __restrict__ b2, float* __restrict__ out)
{
    int tid  = threadIdx.x;
    int lane = tid & 31;
    int wave = tid >> 5;
    int row  = lane & 15;
    int half = lane >> 4;
    int tile = blockIdx.x;
    int jBase = wave * 16;

    const u16* Arow = hmax_bf + (tile * 16 + row) * H;
    const u16* Brow = W2_bf + (jBase + row) * H;

    v8f c = {0.f, 0.f, 0.f, 0.f, 0.f, 0.f, 0.f, 0.f};
    #pragma unroll
    for (int kk = 0; kk < 4; ++kk) {
        int ks = kk * 32;
        ABu a, b;
        a.q[0] = *(const uint4*)(Arow + ks + half * 8);
        a.q[1] = *(const uint4*)(Arow + ks + 16 + half * 8);
        b.q[0] = *(const uint4*)(Brow + ks + half * 8);
        b.q[1] = *(const uint4*)(Brow + ks + 16 + half * 8);
        c = wmma_bf16(a.v, b.v, c);
    }
    int colD = jBase + row;
    float bv = b2[colD];
    #pragma unroll
    for (int v = 0; v < 8; ++v) {
        int m = v + 8 * half;
        out[(tile * 16 + m) * H + colD] = c[v] + bv;
    }
}

// ---------------- host launcher ----------------
extern "C" void kernel_launch(void* const* d_in, const int* in_sizes, int n_in,
                              void* d_out, int out_size, void* d_ws, size_t ws_size,
                              hipStream_t stream) {
    const float* x     = (const float*)d_in[0];
    const int*   esrc  = (const int*)  d_in[1];
    const int*   edst  = (const int*)  d_in[2];
    const float* W1    = (const float*)d_in[3];
    const float* b1    = (const float*)d_in[4];
    const float* W_ih  = (const float*)d_in[5];
    const float* b_ih  = (const float*)d_in[6];
    const float* W_hh  = (const float*)d_in[7];
    const float* b_hh  = (const float*)d_in[8];
    const float* W2    = (const float*)d_in[9];
    const float* b2    = (const float*)d_in[10];
    float* out = (float*)d_out;

    // workspace carve-up (256B aligned)
    char* ws = (char*)d_ws;
    size_t off = 0;
    auto carve = [&](size_t bytes) { void* p = ws + off; off = (off + bytes + 255) & ~(size_t)255; return p; };
    const size_t NH = (size_t)N_NODES * H;
    float* h_f      = (float*)carve(NH * 4);
    float* agg_f    = (float*)carve(NH * 4);
    float* hmax_f   = (float*)carve(NH * 4);
    u16*   h_bf     = (u16*)  carve(NH * 2);
    u16*   agg_bf   = (u16*)  carve(NH * 2);
    u16*   hmax_bf  = (u16*)  carve(NH * 2);
    u16*   Wih_bf   = (u16*)  carve((size_t)TH3 * H * 2);
    u16*   Whh_bf   = (u16*)  carve((size_t)TH3 * H * 2);
    u16*   W2_bf    = (u16*)  carve((size_t)H * H * 2);
    (void)ws_size;

    const int nW = TH3 * H;      // 49152
    const int nW2 = H * H;       // 16384
    const int nNH = (int)NH;     // 3840000

    // 1) weights -> bf16
    cvt_f2bf<<<(nW  + 255) / 256, 256, 0, stream>>>(W_ih, Wih_bf, nW);
    cvt_f2bf<<<(nW  + 255) / 256, 256, 0, stream>>>(W_hh, Whh_bf, nW);
    cvt_f2bf<<<(nW2 + 255) / 256, 256, 0, stream>>>(W2,   W2_bf,  nW2);

    // 2) encoder
    encoder_kernel<<<N_NODES, 128, 0, stream>>>(x, W1, b1, h_f, h_bf, hmax_f, hmax_bf);

    // 3) GRU layers
    for (int l = 0; l < NLAYERS; ++l) {
        zero_f32<<<(nNH + 255) / 256, 256, 0, stream>>>(agg_f, nNH);
        scatter_kernel<<<N_EDGES / 8, 256, 0, stream>>>(h_f, esrc, edst, agg_f);
        cvt_f2bf<<<(nNH + 255) / 256, 256, 0, stream>>>(agg_f, agg_bf, nNH);
        layer_kernel<<<N_NODES / 16, 256, 0, stream>>>(agg_bf, h_bf, Wih_bf, Whh_bf,
                                                       b_ih, b_hh, h_f, hmax_f, hmax_bf);
    }

    // 4) decoder
    decoder_kernel<<<N_NODES / 16, 256, 0, stream>>>(hmax_bf, W2_bf, b2, out);
}
